// Block_74191265071158
// MI455X (gfx1250) — matrix operand order repaired and checked by reading, not compiled, not run on hardware
//
#include <hip/hip_runtime.h>
#include <hip/hip_bf16.h>
#include <math.h>

// ---------------------------------------------------------------------------
// Transformer block for MI455X (gfx1250): bf16 WMMA everywhere, f32 accum.
// Async global->LDS DMA (ASYNCcnt) with double buffering in the GEMMs.
// B=16 N=1024 D=256 H=8 HD=32 IN=256 HID=1024, M = B*N = 16384.
// ---------------------------------------------------------------------------

typedef __attribute__((ext_vector_type(16))) __bf16 v16bf;
typedef __attribute__((ext_vector_type(8)))  float  v8f;

struct B256 { uint4 a, b; };
struct H8   { unsigned short h[8]; };

static __device__ __forceinline__ unsigned short f2bf(float f) {
  unsigned int u = __builtin_bit_cast(unsigned int, f);
  u += 0x7FFFu + ((u >> 16) & 1u);          // round-to-nearest-even
  return (unsigned short)(u >> 16);
}

// Async DMA: 16B global -> LDS per lane, tracked by ASYNCcnt (no VGPR data).
static __device__ __forceinline__ void async_ld16(const void* gptr, void* lptr) {
  const unsigned long long ga = (unsigned long long)(size_t)gptr;
  const unsigned la = (unsigned)(size_t)lptr;   // generic LDS addr low 32 bits
  asm volatile("global_load_async_to_lds_b128 %0, %1, off"
               :: "v"(la), "v"(ga) : "memory");
}
static __device__ __forceinline__ void wait_async() {
  asm volatile("s_wait_asynccnt 0" ::: "memory");
}

// A fragment (16x32 bf16, M x K). ISA 16-bit A layout:
// lanes 0-15: M=lane, K = {0..7, 16..23}; lanes 16-31: same M, K = {8..15, 24..31}
static __device__ __forceinline__ v16bf load_fragA(const unsigned short* lds,
                                                   int row0, int col0, int stride) {
  const int lane = threadIdx.x & 31;
  const int lr = lane & 15, hf = lane >> 4;
  const unsigned short* p = lds + (row0 + lr) * stride + col0 + hf * 8;
  B256 t;
  t.a = *(const uint4*)(p);
  t.b = *(const uint4*)(p + 16);
  return __builtin_bit_cast(v16bf, t);
}

// B fragment (32x16, K x N), source stored transposed as Wt[n][k] (k contiguous).
// lanes 0-15 hold K={0..7,16..23} (N=lane%16), lanes 16-31 hold K={8..15,24..31}.
static __device__ __forceinline__ v16bf load_fragB(const unsigned short* lds,
                                                   int row0, int col0, int stride) {
  const int lane = threadIdx.x & 31;
  const int lr = lane & 15, hf = lane >> 4;
  const unsigned short* p = lds + (row0 + lr) * stride + col0 + hf * 8;
  B256 t;
  t.a = *(const uint4*)(p);
  t.b = *(const uint4*)(p + 16);
  return __builtin_bit_cast(v16bf, t);
}

static __device__ __forceinline__ v8f wmma_bf16(v16bf a, v16bf b, v8f c) {
  return __builtin_amdgcn_wmma_f32_16x16x32_bf16(false, a, false, b,
                                                 (short)0, c, false, false);
}

// ---------------------------------------------------------------------------
// fp32 -> bf16 weight conversion
// ---------------------------------------------------------------------------
__global__ void cvt_bf16_kernel(const float* __restrict__ in,
                                unsigned short* __restrict__ out, int n) {
  int i = blockIdx.x * 256 + threadIdx.x;
  if (i < n) out[i] = f2bf(in[i]);
}

// ---------------------------------------------------------------------------
// LayerNorm over D=256. One wave (32 lanes) per row, 8 elems/lane.
// ---------------------------------------------------------------------------
__global__ __launch_bounds__(256) void ln_kernel(const float* __restrict__ x,
                                                 const float* __restrict__ g,
                                                 const float* __restrict__ b,
                                                 unsigned short* __restrict__ outB,
                                                 float* __restrict__ outHX) {
  const int row  = blockIdx.x * 8 + (threadIdx.x >> 5);
  const int lane = threadIdx.x & 31;
  const float* xr = x + (size_t)row * 256 + lane * 8;
  float4 v0 = *(const float4*)(xr);
  float4 v1 = *(const float4*)(xr + 4);
  float xs[8] = {v0.x, v0.y, v0.z, v0.w, v1.x, v1.y, v1.z, v1.w};

  float s = 0.f, s2 = 0.f;
#pragma unroll
  for (int e = 0; e < 8; ++e) { s += xs[e]; s2 += xs[e] * xs[e]; }
#pragma unroll
  for (int off = 16; off >= 1; off >>= 1) {
    s  += __shfl_xor(s,  off, 32);
    s2 += __shfl_xor(s2, off, 32);
  }
  const float mean = s * (1.0f / 256.0f);
  const float var  = s2 * (1.0f / 256.0f) - mean * mean;
  const float rstd = rsqrtf(var + 1e-5f);

  H8 hb;
  float hx[8];
#pragma unroll
  for (int e = 0; e < 8; ++e) {
    const int col = lane * 8 + e;
    float hn = (xs[e] - mean) * rstd * g[col] + b[col];
    hb.h[e] = f2bf(hn);
    hx[e] = xs[e] + hn;
  }
  *(uint4*)(outB + (size_t)row * 256 + lane * 8) = __builtin_bit_cast(uint4, hb);
  if (outHX) {
    float* o = outHX + (size_t)row * 256 + lane * 8;
    *(float4*)(o)     = make_float4(hx[0], hx[1], hx[2], hx[3]);
    *(float4*)(o + 4) = make_float4(hx[4], hx[5], hx[6], hx[7]);
  }
}

// ---------------------------------------------------------------------------
// Tiled GEMM: C[M,Nc] = act(A[M,K] @ W[Nc,K]^T + bias) [+ res].
// 128x128 block tile, K-step 32, double-buffered async global->LDS DMA.
// 8 waves, each wave: 32x64 tile = 2x4 fragments = 8 WMMA per K-step.
// ---------------------------------------------------------------------------
__global__ __launch_bounds__(256) void gemm_bf16_kernel(
    const unsigned short* __restrict__ A, const unsigned short* __restrict__ W,
    const float* __restrict__ bias, const float* __restrict__ res,
    float* __restrict__ outF, unsigned short* __restrict__ outB,
    int M, int Nc, int K, int gelu) {
  __shared__ unsigned short As[2][128 * 40];   // padded: 80B rows, 16B aligned
  __shared__ unsigned short Ws[2][128 * 40];

  const int t = threadIdx.x;
  const int w = t >> 5, lane = t & 31, lr = lane & 15, hf = lane >> 4;
  const int wm = w & 3, wn = w >> 2;
  const int m0 = blockIdx.x * 128, n0 = blockIdx.y * 128;

  v8f zero = {};
  v8f acc[2][4];
#pragma unroll
  for (int mi = 0; mi < 2; ++mi)
#pragma unroll
    for (int ni = 0; ni < 4; ++ni) acc[mi][ni] = zero;

  // Stage one 128x32 K-slice of A and W into LDS buffer `buf` via async DMA.
  auto stage = [&](int buf, int k0) {
#pragma unroll
    for (int i = 0; i < 2; ++i) {
      const int c = t + 256 * i;
      const int row = c >> 2, ch = (c & 3) * 8;
      async_ld16(&A[(size_t)(m0 + row) * K + k0 + ch], &As[buf][row * 40 + ch]);
      async_ld16(&W[(size_t)(n0 + row) * K + k0 + ch], &Ws[buf][row * 40 + ch]);
    }
  };

  const int nk = K >> 5;
  stage(0, 0);
  wait_async();
  __syncthreads();

  for (int ks = 0; ks < nk; ++ks) {
    const int cur = ks & 1, nxt = cur ^ 1;
    if (ks + 1 < nk) stage(nxt, (ks + 1) * 32);   // DMA overlaps WMMA below

    v16bf af0 = load_fragA(As[cur], wm * 32, 0, 40);
    v16bf af1 = load_fragA(As[cur], wm * 32 + 16, 0, 40);
#pragma unroll
    for (int ni = 0; ni < 4; ++ni) {
      v16bf bf = load_fragB(Ws[cur], wn * 64 + ni * 16, 0, 40);
      acc[0][ni] = wmma_bf16(af0, bf, acc[0][ni]);
      acc[1][ni] = wmma_bf16(af1, bf, acc[1][ni]);
    }
    wait_async();
    __syncthreads();
  }

  // Epilogue: C layout — vgpr v <-> row v + 8*half, column = lane%16.
#pragma unroll
  for (int mi = 0; mi < 2; ++mi)
#pragma unroll
    for (int ni = 0; ni < 4; ++ni)
#pragma unroll
      for (int v = 0; v < 8; ++v) {
        const int row = m0 + wm * 32 + mi * 16 + v + 8 * hf;
        const int col = n0 + wn * 64 + ni * 16 + lr;
        float x = acc[mi][ni][v];
        if (bias) x += bias[col];
        if (gelu) x = 0.5f * x * (1.0f + erff(x * 0.70710678118654752f));
        const size_t idx = (size_t)row * Nc + col;
        if (res)  x += res[idx];
        if (outF) outF[idx] = x;
        if (outB) outB[idx] = f2bf(x);
      }
}

// ---------------------------------------------------------------------------
// Flash-style attention. Grid: (N/128 q-tiles, B*H). Block = 8 waves; each
// wave owns 16 query rows. Online softmax over 8 key tiles of 128.
// qkv is bf16 [B,N,3,H,HD] (i.e. [16384, 768] row-major from the QKV GEMM).
// ---------------------------------------------------------------------------
__global__ __launch_bounds__(256) void attn_kernel(
    const unsigned short* __restrict__ qkv, unsigned short* __restrict__ attnout) {
  __shared__ unsigned short Qs[128 * 40];
  __shared__ unsigned short Ks[128 * 40];
  __shared__ unsigned short Vt[32 * 136];      // V transposed: [d][j]
  __shared__ unsigned short Ps[8 * 16 * 136];  // per-wave P slice [16][128]

  const int t = threadIdx.x, w = t >> 5, lane = t & 31, lr = lane & 15, hf = lane >> 4;
  const int bh = blockIdx.y, bb = bh >> 3, hh = bh & 7;
  const int q0 = blockIdx.x * 128;
  const size_t base = (size_t)bb * 1024 * 768 + hh * 32;

  // Async-stage Q tile [128][32]
#pragma unroll
  for (int i = 0; i < 2; ++i) {
    const int c = t + 256 * i, row = c >> 2, ch = (c & 3) * 8;
    async_ld16(&qkv[base + (size_t)(q0 + row) * 768 + ch], &Qs[row * 40 + ch]);
  }
  wait_async();
  __syncthreads();
  const v16bf qa = load_fragA(Qs, w * 16, 0, 40);

  float m[8], l[8];
  v8f o0 = {}, o1 = {};
#pragma unroll
  for (int v = 0; v < 8; ++v) { m[v] = -1e30f; l[v] = 0.f; }

  for (int jt = 0; jt < 8; ++jt) {
    __syncthreads();  // previous K/Vt consumers done
    const int j0 = jt * 128;
#pragma unroll
    for (int i = 0; i < 2; ++i) {
      const int c = t + 256 * i, row = c >> 2, ch = (c & 3) * 8;
      // K tile: straight async DMA (row-major is already B-operand layout)
      async_ld16(&qkv[base + 256 + (size_t)(j0 + row) * 768 + ch],
                 &Ks[row * 40 + ch]);
      // V tile: manual element transpose into Vt[d][j]
      uint4 vv = *(const uint4*)&qkv[base + 512 + (size_t)(j0 + row) * 768 + ch];
      H8 ve = __builtin_bit_cast(H8, vv);
#pragma unroll
      for (int e = 0; e < 8; ++e) Vt[(ch + e) * 136 + row] = ve.h[e];
    }
    wait_async();
    __syncthreads();

    // S = scale * Q K^T : one WMMA (K=32) per 16x16 tile, 8 tiles per wave
    v8f s[8];
#pragma unroll
    for (int nj = 0; nj < 8; ++nj) {
      v8f z = {};
      v16bf kb = load_fragB(Ks, nj * 16, 0, 40);
      s[nj] = wmma_bf16(qa, kb, z);
    }
    const float scale = 0.17677669529663687f;  // 1/sqrt(32)

    // Row max (reduce across 16-lane group: C layout keeps rows per lane-half)
    float tmax[8];
#pragma unroll
    for (int v = 0; v < 8; ++v) tmax[v] = -1e30f;
#pragma unroll
    for (int nj = 0; nj < 8; ++nj)
#pragma unroll
      for (int v = 0; v < 8; ++v) {
        s[nj][v] *= scale;
        tmax[v] = fmaxf(tmax[v], s[nj][v]);
      }
#pragma unroll
    for (int v = 0; v < 8; ++v)
#pragma unroll
      for (int off = 8; off >= 1; off >>= 1)
        tmax[v] = fmaxf(tmax[v], __shfl_xor(tmax[v], off, 32));

    float al[8];
#pragma unroll
    for (int v = 0; v < 8; ++v) {
      float mn = fmaxf(m[v], tmax[v]);
      al[v] = __expf(m[v] - mn);
      m[v] = mn;
      l[v] *= al[v];
      o0[v] *= al[v];
      o1[v] *= al[v];
    }

    // P = exp(S - m); row sums; spill P (bf16) to LDS to re-shape C -> A layout
    float rs[8] = {0.f, 0.f, 0.f, 0.f, 0.f, 0.f, 0.f, 0.f};
#pragma unroll
    for (int nj = 0; nj < 8; ++nj)
#pragma unroll
      for (int v = 0; v < 8; ++v) {
        float p = __expf(s[nj][v] - m[v]);
        rs[v] += p;
        Ps[(w * 16 + v + 8 * hf) * 136 + nj * 16 + lr] = f2bf(p);
      }
#pragma unroll
    for (int v = 0; v < 8; ++v) {
#pragma unroll
      for (int off = 8; off >= 1; off >>= 1) rs[v] += __shfl_xor(rs[v], off, 32);
      l[v] += rs[v];
    }

    // O += P V : contraction over j in 4 WMMA steps (K=32 each)
#pragma unroll
    for (int jc = 0; jc < 4; ++jc) {
      v16bf pa  = load_fragA(Ps + w * 16 * 136, 0, jc * 32, 136);
      v16bf vb0 = load_fragB(Vt, 0,  jc * 32, 136);
      v16bf vb1 = load_fragB(Vt, 16, jc * 32, 136);
      o0 = wmma_bf16(pa, vb0, o0);
      o1 = wmma_bf16(pa, vb1, o1);
    }
  }

  // Normalize, emit bf16 [16384, 256] (col = h*32 + d) for the proj GEMM
#pragma unroll
  for (int v = 0; v < 8; ++v) {
    const float inv = 1.0f / l[v];
    const int row = bb * 1024 + q0 + w * 16 + v + 8 * hf;
    attnout[(size_t)row * 256 + hh * 32 + lr]      = f2bf(o0[v] * inv);
    attnout[(size_t)row * 256 + hh * 32 + 16 + lr] = f2bf(o1[v] * inv);
  }
}

// ---------------------------------------------------------------------------
// Launch
// ---------------------------------------------------------------------------
extern "C" void kernel_launch(void* const* d_in, const int* in_sizes, int n_in,
                              void* d_out, int out_size, void* d_ws, size_t ws_size,
                              hipStream_t stream) {
  (void)in_sizes; (void)n_in; (void)out_size; (void)ws_size;
  const float* x     = (const float*)d_in[0];
  // d_in[1] = coords (unused by reference)
  const float* g1    = (const float*)d_in[2];
  const float* b1    = (const float*)d_in[3];
  const float* Wqkv  = (const float*)d_in[4];
  const float* Wproj = (const float*)d_in[5];
  const float* bproj = (const float*)d_in[6];
  const float* g2    = (const float*)d_in[7];
  const float* b2    = (const float*)d_in[8];
  const float* W1    = (const float*)d_in[9];
  const float* bb1   = (const float*)d_in[10];
  const float* W2    = (const float*)d_in[11];
  const float* bb2   = (const float*)d_in[12];

  const int M = 16384;
  char* ws = (char*)d_ws;
  size_t off = 0;
  auto carve = [&](size_t bytes) { char* p = ws + off; off += (bytes + 255) & ~(size_t)255; return p; };

  unsigned short* h_bf     = (unsigned short*)carve((size_t)M * 256 * 2);
  float*          hx       = (float*)         carve((size_t)M * 256 * 4);
  unsigned short* wqkv_bf  = (unsigned short*)carve((size_t)768 * 256 * 2);
  unsigned short* wproj_bf = (unsigned short*)carve((size_t)256 * 256 * 2);
  unsigned short* w1_bf    = (unsigned short*)carve((size_t)1024 * 256 * 2);
  unsigned short* w2_bf    = (unsigned short*)carve((size_t)256 * 1024 * 2);
  unsigned short* qkv_bf   = (unsigned short*)carve((size_t)M * 768 * 2);
  unsigned short* att_bf   = (unsigned short*)carve((size_t)M * 256 * 2);
  float*          x2       = (float*)         carve((size_t)M * 256 * 4);
  unsigned short* ln2_bf   = (unsigned short*)carve((size_t)M * 256 * 2);
  unsigned short* m1_bf    = (unsigned short*)carve((size_t)M * 1024 * 2);

  // Weight conversion (fp32 -> bf16)
  cvt_bf16_kernel<<<(768 * 256 + 255) / 256, 256, 0, stream>>>(Wqkv, wqkv_bf, 768 * 256);
  cvt_bf16_kernel<<<(256 * 256 + 255) / 256, 256, 0, stream>>>(Wproj, wproj_bf, 256 * 256);
  cvt_bf16_kernel<<<(1024 * 256 + 255) / 256, 256, 0, stream>>>(W1, w1_bf, 1024 * 256);
  cvt_bf16_kernel<<<(256 * 1024 + 255) / 256, 256, 0, stream>>>(W2, w2_bf, 256 * 1024);

  // h = LN1(x) (bf16), hx = x + h (fp32)
  ln_kernel<<<M / 8, 256, 0, stream>>>(x, g1, b1, h_bf, hx);

  // qkv = h @ Wqkv^T (bf16 out)
  gemm_bf16_kernel<<<dim3(M / 128, 768 / 128), 256, 0, stream>>>(
      h_bf, wqkv_bf, nullptr, nullptr, nullptr, qkv_bf, M, 768, 256, 0);

  // attention -> att_bf [M, 256]
  attn_kernel<<<dim3(8, 128), 256, 0, stream>>>(qkv_bf, att_bf);

  // x2 = att @ Wproj^T + bproj + (x + h)
  gemm_bf16_kernel<<<dim3(M / 128, 256 / 128), 256, 0, stream>>>(
      att_bf, wproj_bf, bproj, hx, x2, nullptr, M, 256, 256, 0);

  // ln2 = LN2(x2) (bf16)
  ln_kernel<<<M / 8, 256, 0, stream>>>(x2, g2, b2, ln2_bf, nullptr);

  // m1 = gelu(ln2 @ W1^T + bb1) (bf16)
  gemm_bf16_kernel<<<dim3(M / 128, 1024 / 128), 256, 0, stream>>>(
      ln2_bf, w1_bf, bb1, nullptr, nullptr, m1_bf, M, 1024, 256, 1);

  // out = m1 @ W2^T + bb2 + x2 (fp32 -> d_out)
  gemm_bf16_kernel<<<dim3(M / 128, 256 / 128), 256, 0, stream>>>(
      m1_bf, w2_bf, bb2, x2, (float*)d_out, nullptr, M, 256, 1024, 0);
}
